// MambaClassifier_2534030705283
// MI455X (gfx1250) — matrix operand order repaired and checked by reading, not compiled, not run on hardware
//
#include <hip/hip_runtime.h>

// ---------------- model constants (match reference) ----------------
#define BB      4
#define LL      2048
#define BL      (BB * LL)          // 8192 rows
#define DMODEL  64
#define ED      128
#define NST     32
#define DCONV   16
#define NLAYERS 4
#define DTRANK  4
#define EPSV    1e-5f

// xproj output (4 + 32 + 32 = 68) padded to 80 for 16-wide WMMA tiles
#define DBC_PAD 80

typedef __bf16 bf16;
typedef __attribute__((ext_vector_type(16))) __bf16 v16bf;
typedef __attribute__((ext_vector_type(8)))  __bf16 v8bf;
typedef __attribute__((ext_vector_type(8)))  float  v8f;

// ---------------- fp32 -> bf16 convert with row zero-padding ----------------
__global__ void k_cvt_pad(const float* __restrict__ src, bf16* __restrict__ dst,
                          int R, int C, int Rp) {
    int idx = blockIdx.x * blockDim.x + threadIdx.x;
    int total = Rp * C;
    if (idx >= total) return;
    int r = idx / C, c = idx - r * C;
    dst[idx] = (r < R) ? (bf16)src[r * C + c] : (bf16)0.0f;
}

// ---------------- RMSNorm: one wave per row of 64, emits bf16 ----------------
__global__ void k_rmsnorm(const float* __restrict__ h, const float* __restrict__ w,
                          bf16* __restrict__ out) {
    int wid  = (blockIdx.x * blockDim.x + threadIdx.x) >> 5;   // row id
    int lane = threadIdx.x & 31;
    if (wid >= BL) return;
    const float* row = h + (size_t)wid * DMODEL;
    float x0 = row[lane], x1 = row[lane + 32];
    float s = x0 * x0 + x1 * x1;
    #pragma unroll
    for (int off = 16; off >= 1; off >>= 1) s += __shfl_xor(s, off, 32);
    float scale = rsqrtf(s * (1.0f / DMODEL) + EPSV);
    bf16* orow = out + (size_t)wid * DMODEL;
    orow[lane]      = (bf16)(x0 * scale * w[lane]);
    orow[lane + 32] = (bf16)(x1 * scale * w[lane + 32]);
}

// ---------------- bf16 WMMA GEMM: C[M,Nc] (+)= A[M,K] * W[Nc,K]^T + bias ----
// One wave per 16x16 output tile. M, Nc multiples of 16; K multiple of 32.
// A,W row-major bf16 (16B-aligned rows). acc_mode: 0 = store, 1 = C += (residual).
__global__ void k_gemm_bf16(const bf16* __restrict__ A, const bf16* __restrict__ W,
                            const float* __restrict__ bias, float* __restrict__ C,
                            int M, int Nc, int K, int ldc, int acc_mode) {
    int wid  = (blockIdx.x * blockDim.x + threadIdx.x) >> 5;
    int lane = threadIdx.x & 31;
    int tilesN = Nc >> 4;
    int tilesM = M >> 4;
    if (wid >= tilesM * tilesN) return;
    int tm = wid / tilesN;
    int tn = wid - tm * tilesN;

    // A fragment addressing (contiguous k order, same as B):
    //  lane<16  -> row M=lane,    K = k0 + 0..15
    //  lane>=16 -> row M=lane-16, K = k0 + 16..31
    int arow = tm * 16 + (lane & 15);
    int aoff = (lane >> 4) * 16;
    // B fragment addressing (same contiguous k order as A): lane holds one column,
    //  lanes 0..15 -> K = k0+0..15, lanes 16..31 -> K = k0+16..31
    int bcol = tn * 16 + (lane & 15);
    int boff = (lane >> 4) * 16;

    const bf16* Arow = A + (size_t)arow * K;
    const bf16* Wrow = W + (size_t)bcol * K;

    v8f acc = {};
    for (int k0 = 0; k0 < K; k0 += 32) {
        if (k0 + 32 < K) __builtin_prefetch(Wrow + k0 + 32, 0, 0); // global_prefetch_b8
        v8bf alo = *(const v8bf*)(Arow + k0 + aoff);
        v8bf ahi = *(const v8bf*)(Arow + k0 + aoff + 8);
        v16bf a;
        #pragma unroll
        for (int i = 0; i < 8; ++i) { a[i] = alo[i]; a[i + 8] = ahi[i]; }
        v16bf b = *(const v16bf*)(Wrow + k0 + boff);
        acc = __builtin_amdgcn_wmma_f32_16x16x32_bf16(
            /*neg_a=*/false, a, /*neg_b=*/false, b,
            /*c_mod=*/(short)0, acc, /*reuse_a=*/false, /*reuse_b=*/false);
    }

    // C fragment: VGPR r -> row tm*16 + r + (lane>=16 ? 8 : 0), col = tn*16 + (lane&15)
    int r0 = tm * 16 + (lane >> 4) * 8;
    float bv = bias ? bias[bcol] : 0.0f;
    #pragma unroll
    for (int r = 0; r < 8; ++r) {
        float* p = C + (size_t)(r0 + r) * ldc + bcol;
        float v = acc[r] + bv;
        if (acc_mode) *p += v; else *p = v;
    }
}

// ---------------- causal depthwise conv (DCONV=16) + SiLU ----------------
// xin = xz[:, 0:128]; writes xc (fp32) and xcb (bf16)
__global__ void k_conv_silu(const float* __restrict__ xz, const float* __restrict__ cw,
                            const float* __restrict__ cb, float* __restrict__ xc,
                            bf16* __restrict__ xcb) {
    int idx = blockIdx.x * blockDim.x + threadIdx.x;
    if (idx >= BL * ED) return;
    int row = idx / ED;           // b*L + t
    int e   = idx - row * ED;
    int b   = row / LL;
    int t   = row - b * LL;
    float accv = cb[e];
    const float* wrow = cw + e * DCONV;
    #pragma unroll
    for (int j = 0; j < DCONV; ++j) {
        int tt = t - (DCONV - 1) + j;
        if (tt >= 0)
            accv += wrow[j] * xz[(size_t)(b * LL + tt) * (2 * ED) + e];
    }
    float s = accv / (1.0f + __expf(-accv));   // silu
    xc[idx]  = s;
    xcb[idx] = (bf16)s;
}

// ---------------- dt projection (K=4) + softplus ----------------
__global__ void k_dtproj(const float* __restrict__ dbc, const float* __restrict__ dtw,
                         const float* __restrict__ dtb, float* __restrict__ delta) {
    int idx = blockIdx.x * blockDim.x + threadIdx.x;
    if (idx >= BL * ED) return;
    int row = idx / ED;
    int e   = idx - row * ED;
    const float* dr = dbc + (size_t)row * DBC_PAD;   // delta_raw at cols [0,4)
    float a = dtb[e];
    #pragma unroll
    for (int r = 0; r < DTRANK; ++r) a += dtw[e * DTRANK + r] * dr[r];
    delta[idx] = (a > 20.0f) ? a : log1pf(__expf(a));  // softplus
}

// ---------------- selective scan: one wave per (b,e); lane n = state index --
__global__ void k_scan(const float* __restrict__ dbc, const float* __restrict__ delta,
                       const float* __restrict__ xc, const float* __restrict__ xz,
                       const float* __restrict__ A_log, const float* __restrict__ Dp,
                       float* __restrict__ y, bf16* __restrict__ yb) {
    int wid  = (blockIdx.x * blockDim.x + threadIdx.x) >> 5;
    int lane = threadIdx.x & 31;                    // n
    if (wid >= BB * ED) return;
    int b = wid / ED;
    int e = wid - b * ED;
    float Aen = -__expf(A_log[e * NST + lane]);
    float Dpe = Dp[e];
    float hst = 0.0f;
    for (int t = 0; t < LL; ++t) {
        int row = b * LL + t;
        float dlt = delta[(size_t)row * ED + e];
        float xcv = xc[(size_t)row * ED + e];
        const float* dr = dbc + (size_t)row * DBC_PAD;
        float Bn = dr[DTRANK + lane];               // cols [4,36)
        float Cn = dr[DTRANK + NST + lane];         // cols [36,68)
        float dA = __expf(dlt * Aen);
        hst = dA * hst + dlt * Bn * xcv;
        float contrib = hst * Cn;
        #pragma unroll
        for (int off = 16; off >= 1; off >>= 1) contrib += __shfl_xor(contrib, off, 32);
        if (lane == 0) {
            float zv = xz[(size_t)row * (2 * ED) + ED + e];
            float sz = zv / (1.0f + __expf(-zv));
            float yv = (contrib + Dpe * xcv) * sz;
            y[(size_t)row * ED + e]  = yv;
            yb[(size_t)row * ED + e] = (bf16)yv;
        }
    }
}

// ---------------- final classifier head: out[b] = y[b,L-1,:] . fc_w + fc_b --
__global__ void k_fc(const float* __restrict__ y, const float* __restrict__ fcw,
                     const float* __restrict__ fcb, float* __restrict__ out) {
    __shared__ float red[128];
    int b = blockIdx.x, e = threadIdx.x;
    float v = y[(size_t)(b * LL + LL - 1) * ED + e] * fcw[e];
    red[e] = v;
    __syncthreads();
    for (int s = 64; s >= 1; s >>= 1) {
        if (e < s) red[e] += red[e + s];
        __syncthreads();
    }
    if (e == 0) out[b] = red[0] + fcb[0];
}

// ---------------- host orchestration ----------------
extern "C" void kernel_launch(void* const* d_in, const int* in_sizes, int n_in,
                              void* d_out, int out_size, void* d_ws, size_t ws_size,
                              hipStream_t stream) {
    (void)in_sizes; (void)n_in; (void)out_size; (void)ws_size;
    const float* x        = (const float*)d_in[0];
    const float* in_w     = (const float*)d_in[1];
    const float* in_b     = (const float*)d_in[2];
    const float* conv_w   = (const float*)d_in[3];
    const float* conv_b   = (const float*)d_in[4];
    const float* xproj_w  = (const float*)d_in[5];
    const float* dtproj_w = (const float*)d_in[6];
    const float* dtproj_b = (const float*)d_in[7];
    const float* A_log    = (const float*)d_in[8];
    const float* Dp       = (const float*)d_in[9];
    const float* outproj_w= (const float*)d_in[10];
    const float* outproj_b= (const float*)d_in[11];
    const float* norm_w   = (const float*)d_in[12];
    const float* fc_w     = (const float*)d_in[13];
    const float* fc_b     = (const float*)d_in[14];
    float* out = (float*)d_out;

    // workspace carve-up (256B aligned)
    char* base = (char*)d_ws;
    size_t off = 0;
    auto carve = [&](size_t bytes) -> char* {
        char* p = base + off;
        off += (bytes + 255) & ~(size_t)255;
        return p;
    };
    float* h     = (float*)carve((size_t)BL * DMODEL * 4);
    bf16*  hn    = (bf16*) carve((size_t)BL * DMODEL * 2);
    float* xz    = (float*)carve((size_t)BL * 2 * ED * 4);
    float* xc    = (float*)carve((size_t)BL * ED * 4);
    bf16*  xcb   = (bf16*) carve((size_t)BL * ED * 2);
    float* dbc   = (float*)carve((size_t)BL * DBC_PAD * 4);
    float* delta = (float*)carve((size_t)BL * ED * 4);
    float* y     = (float*)carve((size_t)BL * ED * 4);
    bf16*  yb    = (bf16*) carve((size_t)BL * ED * 2);
    bf16*  w_in  = (bf16*) carve((size_t)(2 * ED) * DMODEL * 2);
    bf16*  w_xp  = (bf16*) carve((size_t)DBC_PAD * ED * 2);
    bf16*  w_op  = (bf16*) carve((size_t)DMODEL * ED * 2);

    // residual stream starts as input x
    hipMemcpyAsync(h, x, (size_t)BL * DMODEL * 4, hipMemcpyDeviceToDevice, stream);

    const int TPB = 128;
    for (int l = 0; l < NLAYERS; ++l) {
        const float* in_w_l   = in_w     + (size_t)l * 2 * ED * DMODEL;
        const float* in_b_l   = in_b     + (size_t)l * 2 * ED;
        const float* conv_w_l = conv_w   + (size_t)l * ED * DCONV;
        const float* conv_b_l = conv_b   + (size_t)l * ED;
        const float* xproj_l  = xproj_w  + (size_t)l * (DTRANK + 2 * NST) * ED;
        const float* dtw_l    = dtproj_w + (size_t)l * ED * DTRANK;
        const float* dtb_l    = dtproj_b + (size_t)l * ED;
        const float* Alog_l   = A_log    + (size_t)l * ED * NST;
        const float* Dp_l     = Dp       + (size_t)l * ED;
        const float* opw_l    = outproj_w+ (size_t)l * DMODEL * ED;
        const float* opb_l    = outproj_b+ (size_t)l * DMODEL;
        const float* nw_l     = norm_w   + (size_t)l * DMODEL;

        // 1) RMSNorm -> bf16
        k_rmsnorm<<<(BL * 32 + TPB - 1) / TPB, TPB, 0, stream>>>(h, nw_l, hn);

        // 2) in_proj GEMM: [BL,64] x [256,64]^T -> xz [BL,256]
        k_cvt_pad<<<(2 * ED * DMODEL + 255) / 256, 256, 0, stream>>>(
            in_w_l, w_in, 2 * ED, DMODEL, 2 * ED);
        {
            int waves = (BL / 16) * ((2 * ED) / 16);
            k_gemm_bf16<<<(waves * 32 + TPB - 1) / TPB, TPB, 0, stream>>>(
                hn, w_in, in_b_l, xz, BL, 2 * ED, DMODEL, 2 * ED, 0);
        }

        // 3) causal depthwise conv + SiLU
        k_conv_silu<<<(BL * ED + 255) / 256, 256, 0, stream>>>(
            xz, conv_w_l, conv_b_l, xc, xcb);

        // 4) x_proj GEMM: [BL,128] x [80(pad 68),128]^T -> dbc [BL,80]
        k_cvt_pad<<<(DBC_PAD * ED + 255) / 256, 256, 0, stream>>>(
            xproj_l, w_xp, DTRANK + 2 * NST, ED, DBC_PAD);
        {
            int waves = (BL / 16) * (DBC_PAD / 16);
            k_gemm_bf16<<<(waves * 32 + TPB - 1) / TPB, TPB, 0, stream>>>(
                xcb, w_xp, nullptr, dbc, BL, DBC_PAD, ED, DBC_PAD, 0);
        }

        // 5) dt projection (K=4) + softplus
        k_dtproj<<<(BL * ED + 255) / 256, 256, 0, stream>>>(dbc, dtw_l, dtb_l, delta);

        // 6) selective scan + skip + gate (one wave per (b,e); lane = state n)
        k_scan<<<(BB * ED * 32 + TPB - 1) / TPB, TPB, 0, stream>>>(
            dbc, delta, xc, xz, Alog_l, Dp_l, y, yb);

        if (l < NLAYERS - 1) {
            // 7) out_proj GEMM fused with residual: h += y @ outproj_w^T + b
            k_cvt_pad<<<(DMODEL * ED + 255) / 256, 256, 0, stream>>>(
                opw_l, w_op, DMODEL, ED, DMODEL);
            int waves = (BL / 16) * (DMODEL / 16);
            k_gemm_bf16<<<(waves * 32 + TPB - 1) / TPB, TPB, 0, stream>>>(
                yb, w_op, opb_l, h, BL, DMODEL, ED, DMODEL, 1);
        } else {
            // 8) classifier head on y[:, L-1, :]
            k_fc<<<BB, ED, 0, stream>>>(y, fc_w, fc_b, out);
        }
    }
}
